// MultiHeadedAttention_33835752358170
// MI455X (gfx1250) — matrix operand order repaired and checked by reading, not compiled, not run on hardware
//
#include <hip/hip_runtime.h>
#include <hip/hip_bf16.h>

typedef __attribute__((ext_vector_type(16))) _Float16 v16h;
typedef __attribute__((ext_vector_type(8)))  _Float16 v8h;
typedef __attribute__((ext_vector_type(4)))  _Float16 v4h;
typedef __attribute__((ext_vector_type(8)))  float    v8f;

#define WMMA_F16(a, b, c) \
    __builtin_amdgcn_wmma_f32_16x16x32_f16(false, (a), false, (b), (short)0, (c), false, false)

constexpr int Bn   = 4;
constexpr int Sn   = 2048;
constexpr int DM   = 1024;
constexpr int QKVW = 192;   // 3 * d_k
constexpr int NH   = 16;
constexpr int HD   = 4;     // per-head dim
constexpr int DKw  = 64;    // d_k
constexpr int Mrows = Bn * Sn; // 8192

// ---------------- f32 -> f16 convert ----------------
__global__ __launch_bounds__(256) void cvt_f16(const float* __restrict__ src,
                                               _Float16* __restrict__ dst, int n) {
    int i = blockIdx.x * 256 + threadIdx.x;
    if (i < n) dst[i] = (_Float16)src[i];
}

// ---------------- QKV projection: [8192 x 1024] x [1024 x 192] + bias ----------------
// One 16x16 output tile per wave. K stepped by 32 with v_wmma_f32_16x16x32_f16.
__global__ __launch_bounds__(256) void qkv_gemm(const _Float16* __restrict__ xh,
                                                const _Float16* __restrict__ wh,
                                                const float* __restrict__ bias,
                                                float* __restrict__ qkv) {
    int wid  = (blockIdx.x * 256 + threadIdx.x) >> 5;
    int lane = threadIdx.x & 31;
    int half = lane >> 4, lm = lane & 15;
    int tm = (wid / (QKVW / 16)) * 16;
    int tn = (wid % (QKVW / 16)) * 16;

    const _Float16* arow = xh + (size_t)(tm + lm) * DM;
    const _Float16* brow = wh + (size_t)(tn + lm) * DM;

    v8f c = {};
    for (int kk = 0; kk < DM; kk += 32) {
        // A layout: element j -> K = 16*half + j (contiguous 32B per lane)
        v16h a = *(const v16h*)(arow + kk + 16 * half);
        // B layout: element j -> K = 16*half + j (contiguous 32B per lane)
        v16h b = *(const v16h*)(brow + kk + 16 * half);
        c = WMMA_F16(a, b, c);
    }
    float bv = bias[tn + lm];
    #pragma unroll
    for (int r = 0; r < 8; ++r) {
        int m = tm + r + 8 * half;
        qkv[(size_t)m * QKVW + tn + lm] = c[r] + bv;
    }
}

// ---------------- Flash attention per (b,h), HEAD_DIM = 4 ----------------
__global__ __launch_bounds__(256) void attn_kernel(const float* __restrict__ qkv,
                                                   _Float16* __restrict__ attn_h) {
    constexpr int SPLIT = 8;
    int bh   = blockIdx.x / SPLIT;
    int part = blockIdx.x % SPLIT;
    int b = bh >> 4, h = bh & 15;

    __shared__ _Float16 sk[Sn * HD];        // 16 KB  K[l][d]   (f16)
    __shared__ _Float16 svt[HD * Sn];       // 16 KB  V^T[d][l] (f16)
    __shared__ _Float16 sp[8 * 16 * 32];    //  8 KB  per-wave P scratch

    int tid = threadIdx.x;
    for (int idx = tid; idx < Sn * HD; idx += 256) {
        int l = idx >> 2, d = idx & 3;
        size_t base = (size_t)(b * Sn + l) * QKVW + h * HD + d;
        sk[idx]         = (_Float16)qkv[base + DKw];        // k
        svt[d * Sn + l] = (_Float16)qkv[base + 2 * DKw];    // v (transposed)
    }
    __syncthreads();

    int wave = tid >> 5, lane = tid & 31;
    int half = lane >> 4, lm = lane & 15;
    _Float16* myp = sp + wave * (16 * 32);

    const int tilesPerPart = (Sn / 16) / SPLIT;  // 16
    const int tilesPerWave = tilesPerPart / 8;   // 2

    for (int t = 0; t < tilesPerWave; ++t) {
        int stile = part * tilesPerPart + wave * tilesPerWave + t;
        int s0 = stile * 16;

        // A = q tile, pre-scaled by 1/sqrt(d_k)=1/8, zero-padded K (only K<4 live)
        v16h aq = {};
        if (half == 0) {
            const float* qp = qkv + (size_t)(b * Sn + s0 + lm) * QKVW + h * HD;
            #pragma unroll
            for (int j = 0; j < HD; ++j) aq[j] = (_Float16)(qp[j] * 0.125f);
        }

        v8f acc = {};
        float mrow[8], lrow[8];
        #pragma unroll
        for (int r = 0; r < 8; ++r) { mrow[r] = -3.0e38f; lrow[r] = 0.f; }

        for (int lc = 0; lc < Sn; lc += 32) {
            // B = k^T columns lc..lc+31, two 16-wide WMMAs (K padded, only K<4 live)
            v16h bk0 = {}, bk1 = {};
            if (half == 0) {
                v4h k0 = *(const v4h*)(sk + (lc + lm) * HD);
                v4h k1 = *(const v4h*)(sk + (lc + 16 + lm) * HD);
                #pragma unroll
                for (int j = 0; j < HD; ++j) { bk0[j] = k0[j]; bk1[j] = k1[j]; }
            }
            v8f z = {};
            v8f sc0 = WMMA_F16(aq, bk0, z);
            v8f sc1 = WMMA_F16(aq, bk1, z);

            float p0[8], p1[8];
            #pragma unroll
            for (int r = 0; r < 8; ++r) {
                float tmx = fmaxf(sc0[r], sc1[r]);
                tmx = fmaxf(tmx, __shfl_xor(tmx, 1, 32));
                tmx = fmaxf(tmx, __shfl_xor(tmx, 2, 32));
                tmx = fmaxf(tmx, __shfl_xor(tmx, 4, 32));
                tmx = fmaxf(tmx, __shfl_xor(tmx, 8, 32));
                float newm = fmaxf(mrow[r], tmx);
                float rs   = __expf(mrow[r] - newm);   // rescale factor
                mrow[r] = newm;
                acc[r] *= rs;
                lrow[r] *= rs;
                p0[r] = __expf(sc0[r] - newm);
                p1[r] = __expf(sc1[r] - newm);
                float sum = p0[r] + p1[r];
                sum += __shfl_xor(sum, 1, 32);
                sum += __shfl_xor(sum, 2, 32);
                sum += __shfl_xor(sum, 4, 32);
                sum += __shfl_xor(sum, 8, 32);
                lrow[r] += sum;
                // D layout -> LDS row-major P tile [16][32]
                myp[(r + 8 * half) * 32 + lm]      = (_Float16)p0[r];
                myp[(r + 8 * half) * 32 + 16 + lm] = (_Float16)p1[r];
            }
            asm volatile("" ::: "memory");   // keep LDS store->load order (same-wave, in-order DS)

            // Reload P in A layout
            v16h ap = *(const v16h*)(myp + lm * 32 + 16 * half);

            // B = V chunk [32 x 16], only n<4 columns live
            v16h bvv = {};
            if (lm < HD) bvv = *(const v16h*)(svt + lm * Sn + lc + 16 * half);

            acc = WMMA_F16(ap, bvv, acc);
        }

        if (lm < HD) {
            #pragma unroll
            for (int r = 0; r < 8; ++r) {
                int s = s0 + r + 8 * half;
                attn_h[(size_t)(b * Sn + s) * DKw + h * HD + lm] =
                    (_Float16)(acc[r] / lrow[r]);
            }
        }
    }
}

// ---------------- Output projection: [8192 x 64] x [64 x 1024] + bias ----------------
__global__ __launch_bounds__(256) void out_gemm(const _Float16* __restrict__ ah,
                                                const _Float16* __restrict__ wh,
                                                const float* __restrict__ bias,
                                                float* __restrict__ out) {
    int wid  = (blockIdx.x * 256 + threadIdx.x) >> 5;
    int lane = threadIdx.x & 31;
    int half = lane >> 4, lm = lane & 15;
    int tm = (wid >> 6) * 16;   // 1024/16 = 64 N-tiles
    int tn = (wid & 63) * 16;

    const _Float16* arow = ah + (size_t)(tm + lm) * DKw;
    const _Float16* brow = wh + (size_t)(tn + lm) * DKw;

    v8f c = {};
    #pragma unroll
    for (int kk = 0; kk < DKw; kk += 32) {
        v16h a = *(const v16h*)(arow + kk + 16 * half);
        v16h b = *(const v16h*)(brow + kk + 16 * half);
        c = WMMA_F16(a, b, c);
    }
    float bv = bias[tn + lm];
    #pragma unroll
    for (int r = 0; r < 8; ++r) {
        int m = tm + r + 8 * half;
        out[(size_t)m * DM + tn + lm] = c[r] + bv;
    }
}

extern "C" void kernel_launch(void* const* d_in, const int* in_sizes, int n_in,
                              void* d_out, int out_size, void* d_ws, size_t ws_size,
                              hipStream_t stream) {
    const float* x     = (const float*)d_in[0];
    const float* w_in  = (const float*)d_in[1];
    const float* b_in  = (const float*)d_in[2];
    const float* w_out = (const float*)d_in[3];
    const float* b_out = (const float*)d_in[4];
    float* out = (float*)d_out;

    char* ws = (char*)d_ws;
    _Float16* xh    = (_Float16*)(ws);                        // 8192*1024 f16 = 16 MB
    _Float16* winh  = (_Float16*)(ws + 16777216);             // 192*1024  f16
    _Float16* wouth = (_Float16*)(ws + 17170432);             // 1024*64   f16
    float*    qkv   = (float*)   (ws + 17301504);             // 8192*192  f32
    _Float16* ah    = (_Float16*)(ws + 23592960);             // 8192*64   f16

    cvt_f16<<<(Mrows * DM + 255) / 256, 256, 0, stream>>>(x, xh, Mrows * DM);
    cvt_f16<<<(QKVW * DM + 255) / 256, 256, 0, stream>>>(w_in, winh, QKVW * DM);
    cvt_f16<<<(DM * DKw + 255) / 256, 256, 0, stream>>>(w_out, wouth, DM * DKw);

    // 512 M-tiles * 12 N-tiles = 6144 waves / 8 waves per block
    qkv_gemm<<<768, 256, 0, stream>>>(xh, winh, b_in, qkv);

    // B*H*SPLIT = 4*16*8 = 512 blocks
    attn_kernel<<<512, 256, 0, stream>>>(qkv, ah);

    // 512 M-tiles * 64 N-tiles = 32768 waves / 8 per block
    out_gemm<<<4096, 256, 0, stream>>>(ah, wouth, b_out, out);
}
